// VeroneseDecoding_7765300871194
// MI455X (gfx1250) — matrix operand order repaired and checked by reading, not compiled, not run on hardware
//
#include <hip/hip_runtime.h>
#include <hip/hip_bf16.h>

typedef __attribute__((ext_vector_type(16))) _Float16 v16h;
typedef __attribute__((ext_vector_type(8)))  float    v8f;
typedef __attribute__((ext_vector_type(8)))  unsigned v8u;

#define IN_F  64
#define HID   256
#define OUT_F 64
#define HSTR  258   // halves per h-row in LDS (padded: 129 dwords -> bank-friendly)
#define MLP_WAVES 4

static __device__ __forceinline__ unsigned pack2_f16(float a, float b) {
    _Float16 ha = (_Float16)a, hb = (_Float16)b;
    unsigned short ua = __builtin_bit_cast(unsigned short, ha);
    unsigned short ub = __builtin_bit_cast(unsigned short, hb);
    return ((unsigned)ub << 16) | (unsigned)ua;
}

// ---------------------------------------------------------------------------
// Kernel 1: convert W1 (64x256) / W2 (256x60, padded to 64) to f16 and pack
// them into CDNA5 WMMA B-fragment layout:
//   fragment (per 16x16x32 step) = 32 lanes x 8 dwords, lane l: n = l&15,
//   kbase = (l<16 ? 0 : 8), dword v holds halves (K, K+1), K = kbase+2v (v<4)
//   or kbase+16+2(v-4) (v>=4).
// w1p: [ntile(16)][kstep(2)][lane(32)][v(8)]  = 8192 dwords
// w2p: [ntile(4)][kstep(8)][lane(32)][v(8)]   = 8192 dwords
// ---------------------------------------------------------------------------
__global__ void pack_weights_kernel(const float* __restrict__ W1,
                                    const float* __restrict__ W2,
                                    unsigned* __restrict__ w1p,
                                    unsigned* __restrict__ w2p) {
    int t = blockIdx.x * blockDim.x + threadIdx.x;   // 0..8191
    if (t >= 8192) return;
    int v = t & 7;
    int l = (t >> 3) & 31;
    int f = t >> 8;                                  // 0..31 fragment id
    int nlane = l & 15;
    int khalf = ((l < 16) ? 0 : 8) + ((v < 4) ? 2 * v : 16 + 2 * (v - 4));

    // W1 fragment: f = j*2 + s  (j: 16 n-tiles of HID, s: 2 k-steps of IN_F)
    {
        int j = f >> 1, s = f & 1;
        int n  = j * 16 + nlane;
        int k0 = s * 32 + khalf;
        w1p[t] = pack2_f16(W1[k0 * HID + n], W1[(k0 + 1) * HID + n]);
    }
    // W2 fragment: f = tt*8 + ss (tt: 4 n-tiles of 64-padded, ss: 8 k-steps of HID)
    {
        int tt = f >> 3, ss = f & 7;
        int n  = tt * 16 + nlane;
        int k0 = ss * 32 + khalf;
        float x0 = (n < 60) ? W2[k0 * 60 + n]       : 0.0f;
        float x1 = (n < 60) ? W2[(k0 + 1) * 60 + n] : 0.0f;
        w2p[t] = pack2_f16(x0, x1);
    }
}

// ---------------------------------------------------------------------------
// Kernel 2: quaternion recovery. One thread per row: build symmetric 4x4 from
// z[:10] (upper-tri order), cyclic Jacobi (6 sweeps, fully unrolled, all in
// registers), pick eigenvector of the largest eigenvalue, fix sign of q0.
// Writes out[:, 0:4].
// ---------------------------------------------------------------------------
__global__ void quat_kernel(const float* __restrict__ z,
                            float* __restrict__ out, int nrows) {
    int i = blockIdx.x * blockDim.x + threadIdx.x;
    if (i >= nrows) return;
    const float* zr = z + (long)i * IN_F;

    float A[4][4], V[4][4];
    A[0][0] = zr[0];
    A[0][1] = A[1][0] = zr[1];
    A[0][2] = A[2][0] = zr[2];
    A[0][3] = A[3][0] = zr[3];
    A[1][1] = zr[4];
    A[1][2] = A[2][1] = zr[5];
    A[1][3] = A[3][1] = zr[6];
    A[2][2] = zr[7];
    A[2][3] = A[3][2] = zr[8];
    A[3][3] = zr[9];
#pragma unroll
    for (int r = 0; r < 4; ++r)
#pragma unroll
        for (int c = 0; c < 4; ++c) V[r][c] = (r == c) ? 1.0f : 0.0f;

    const int P[6] = {0, 0, 0, 1, 1, 2};
    const int Q[6] = {1, 2, 3, 2, 3, 3};
#pragma unroll
    for (int sweep = 0; sweep < 6; ++sweep) {
#pragma unroll
        for (int rot = 0; rot < 6; ++rot) {
            const int p = P[rot], q = Q[rot];
            float apq = A[p][q];
            if (__builtin_fabsf(apq) > 1e-20f) {
                float theta = 0.5f * (A[q][q] - A[p][p]) / apq;
                float tval  = __builtin_copysignf(1.0f, theta) /
                              (__builtin_fabsf(theta) + __builtin_sqrtf(theta * theta + 1.0f));
                float cc = 1.0f / __builtin_sqrtf(tval * tval + 1.0f);
                float ss = tval * cc;
                // A := A * J   (columns p,q)
#pragma unroll
                for (int r = 0; r < 4; ++r) {
                    float arp = A[r][p], arq = A[r][q];
                    A[r][p] = cc * arp - ss * arq;
                    A[r][q] = ss * arp + cc * arq;
                }
                // A := J^T * A (rows p,q)
#pragma unroll
                for (int r = 0; r < 4; ++r) {
                    float apr = A[p][r], aqr = A[q][r];
                    A[p][r] = cc * apr - ss * aqr;
                    A[q][r] = ss * apr + cc * aqr;
                }
                // V := V * J
#pragma unroll
                for (int r = 0; r < 4; ++r) {
                    float vrp = V[r][p], vrq = V[r][q];
                    V[r][p] = cc * vrp - ss * vrq;
                    V[r][q] = ss * vrp + cc * vrq;
                }
            }
        }
    }

    float best = A[0][0];
    float q0 = V[0][0], q1 = V[1][0], q2 = V[2][0], q3 = V[3][0];
#pragma unroll
    for (int k = 1; k < 4; ++k) {
        if (A[k][k] > best) {
            best = A[k][k];
            q0 = V[0][k]; q1 = V[1][k]; q2 = V[2][k]; q3 = V[3][k];
        }
    }
    float sgn = (q0 < 0.0f) ? -1.0f : 1.0f;   // sign(0) -> +1 per reference
    float* o = out + (long)i * OUT_F;
    o[0] = q0 * sgn; o[1] = q1 * sgn; o[2] = q2 * sgn; o[3] = q3 * sgn;
}

// ---------------------------------------------------------------------------
// Kernel 3: fused MLP via WMMA. 4 waves/block, one 16-row tile per wave.
// Stage 1: h(16x256) = relu(z(16x64) @ W1)  -> LDS (f16, padded rows)
// Stage 2: out(16x64pad) = h @ W2           -> global (cols 4..63)
// A fragments follow the 16-bit A layout: lane l: m=l&15, kb=(l<16?0:8),
// dword v: K = kb+2v (v<4) or kb+16+2(v-4) (v>=4).
// ---------------------------------------------------------------------------
__global__ __launch_bounds__(MLP_WAVES * 32)
void mlp_kernel(const float* __restrict__ z,
                const float* __restrict__ b1,
                const float* __restrict__ b2,
                const unsigned* __restrict__ w1p,
                const unsigned* __restrict__ w2p,
                float* __restrict__ out) {
    __shared__ _Float16 hlds[MLP_WAVES][16 * HSTR];

    const int lane = threadIdx.x & 31;
    const int wave = threadIdx.x >> 5;
    const int m    = lane & 15;
    const int hi   = lane >> 4;          // 0 for lanes 0-15, 1 for 16-31
    const long row0 = ((long)blockIdx.x * MLP_WAVES + wave) * 16;

    // ---- load z A-fragments (K=0..31 and K=32..63) straight from global ----
    const float* zrow = z + (row0 + m) * IN_F;
    v16h a0, a1;
#pragma unroll
    for (int v = 0; v < 8; ++v) {
        int k = (v < 4) ? (hi * 8 + 2 * v) : (hi * 8 + 16 + 2 * (v - 4));
        float2 f0 = *(const float2*)(zrow + k);
        float2 f1 = *(const float2*)(zrow + 32 + k);
        a0[2 * v]     = (_Float16)f0.x;
        a0[2 * v + 1] = (_Float16)f0.y;
        a1[2 * v]     = (_Float16)f1.x;
        a1[2 * v + 1] = (_Float16)f1.y;
    }

    // ---- stage 1: 16 n-tiles of HID, K = 64 (2 WMMAs each) ----
    _Float16* hbase = &hlds[wave][0];
    for (int j = 0; j < 16; ++j) {
        float bias = b1[j * 16 + m];
        v8f c;
#pragma unroll
        for (int v = 0; v < 8; ++v) c[v] = bias;

        v8u u0 = *(const v8u*)(w1p + ((j * 2 + 0) * 32 + lane) * 8);
        v8u u1 = *(const v8u*)(w1p + ((j * 2 + 1) * 32 + lane) * 8);
        c = __builtin_amdgcn_wmma_f32_16x16x32_f16(
            false, a0, false, __builtin_bit_cast(v16h, u0), (short)0, c, false, false);
        c = __builtin_amdgcn_wmma_f32_16x16x32_f16(
            false, a1, false, __builtin_bit_cast(v16h, u1), (short)0, c, false, false);

        // relu + scatter into LDS h tile (C/D layout: n = 16j+m, mrow = v+8*hi)
#pragma unroll
        for (int v = 0; v < 8; ++v) {
            float r = c[v] > 0.0f ? c[v] : 0.0f;
            int mrow = v + 8 * hi;
            hbase[mrow * HSTR + j * 16 + m] = (_Float16)r;
        }
    }

    __syncthreads();

    // ---- stage 2: 4 n-tiles of padded OUT (64), K = 256 (8 WMMAs each) ----
    for (int t = 0; t < 4; ++t) {
        int n = t * 16 + m;
        float bias = (n < 60) ? b2[n] : 0.0f;
        v8f c;
#pragma unroll
        for (int v = 0; v < 8; ++v) c[v] = bias;

        for (int s = 0; s < 8; ++s) {
            // A fragment of h from LDS
            v16h a;
            int kb = s * 32 + hi * 8;
#pragma unroll
            for (int v = 0; v < 8; ++v) {
                int k = (v < 4) ? (kb + 2 * v) : (kb + 16 + 2 * (v - 4));
                unsigned d = *(const unsigned*)&hbase[m * HSTR + k];
                a[2 * v]     = __builtin_bit_cast(_Float16, (unsigned short)(d & 0xffffu));
                a[2 * v + 1] = __builtin_bit_cast(_Float16, (unsigned short)(d >> 16));
            }
            v8u u = *(const v8u*)(w2p + ((t * 8 + s) * 32 + lane) * 8);
            c = __builtin_amdgcn_wmma_f32_16x16x32_f16(
                false, a, false, __builtin_bit_cast(v16h, u), (short)0, c, false, false);
        }

        if (n < 60) {
#pragma unroll
            for (int v = 0; v < 8; ++v) {
                int mrow = v + 8 * hi;
                out[(row0 + mrow) * OUT_F + 4 + n] = c[v];
            }
        }
    }
}

// ---------------------------------------------------------------------------
extern "C" void kernel_launch(void* const* d_in, const int* in_sizes, int n_in,
                              void* d_out, int out_size, void* d_ws, size_t ws_size,
                              hipStream_t stream) {
    const float* z  = (const float*)d_in[0];
    const float* W1 = (const float*)d_in[1];
    const float* b1 = (const float*)d_in[2];
    const float* W2 = (const float*)d_in[3];
    const float* b2 = (const float*)d_in[4];
    float* out = (float*)d_out;

    const int nrows = in_sizes[0] / IN_F;          // 262144

    // workspace: 8192 dwords packed W1 + 8192 dwords packed W2 (64 KB total)
    unsigned* w1p = (unsigned*)d_ws;
    unsigned* w2p = w1p + 8192;

    pack_weights_kernel<<<32, 256, 0, stream>>>(W1, W2, w1p, w2p);

    quat_kernel<<<(nrows + 255) / 256, 256, 0, stream>>>(z, out, nrows);

    const int tiles  = nrows / 16;                 // 16384
    const int blocks = tiles / MLP_WAVES;          // 4096
    mlp_kernel<<<blocks, MLP_WAVES * 32, 0, stream>>>(z, b1, b2, w1p, w2p, out);
}
